// CausalSelfAttention_34943853920401
// MI455X (gfx1250) — compile-verified, matrix operand order checked by reading, not run on hardware
//
#include <hip/hip_runtime.h>
#include <hip/hip_bf16.h>
#include <math.h>
#include <stdint.h>

#define NB     4
#define LSEQ   2048
#define DMODEL 512
#define NHD    8
#define HDIM   64

typedef __attribute__((ext_vector_type(16))) __bf16 bf16x16;
typedef __attribute__((ext_vector_type(8)))  __bf16 bf16x8;
typedef __attribute__((ext_vector_type(8)))  float  f32x8;
typedef int v4i __attribute__((vector_size(4 * sizeof(int))));  // matches builtin param

#define AS1 __attribute__((address_space(1)))
#define AS3 __attribute__((address_space(3)))

__device__ __forceinline__ f32x8 wmma_bf16(bf16x16 a, bf16x16 b, f32x8 c) {
  return __builtin_amdgcn_wmma_f32_16x16x32_bf16(false, a, false, b, (short)0, c,
                                                 false, false);
}

__device__ __forceinline__ f32x8 zero8() {
  f32x8 z = {0.f, 0.f, 0.f, 0.f, 0.f, 0.f, 0.f, 0.f};
  return z;
}

// Per-lane fragment from a contiguous bf16 row (A 16x32 / B 32x16 packing):
// elements e<8 -> k = koff+8g+e, e>=8 -> k = koff+16+8g+(e-8).
__device__ __forceinline__ bf16x16 frag_from_bf16_row(const __bf16* row, int koff, int g) {
  bf16x8 lo = *reinterpret_cast<const bf16x8*>(row + koff + 8 * g);
  bf16x8 hi = *reinterpret_cast<const bf16x8*>(row + koff + 16 + 8 * g);
  bf16x16 r;
#pragma unroll
  for (int e = 0; e < 8; ++e) { r[e] = lo[e]; r[8 + e] = hi[e]; }
  return r;
}

// ---- gfx1250 async global->LDS copy (16B per lane) + ASYNCcnt waits --------
__device__ __forceinline__ void async_cp_b128(const __bf16* g, __bf16* l) {
#if __has_builtin(__builtin_amdgcn_global_load_async_to_lds_b128)
  // Signature (from clang diag): (v4i AS1*, v4i AS3*, imm offset, imm cpol).
  // LDS aperture: low 32 bits of the flat address are the LDS byte offset.
  __builtin_amdgcn_global_load_async_to_lds_b128(
      (AS1 v4i*)(uintptr_t)g, (AS3 v4i*)(uintptr_t)l, 0, 0);
#else
  asm volatile("global_load_async_to_lds_b128 %0, %1, off"
               :: "v"((uint32_t)(uintptr_t)l), "v"(g) : "memory");
#endif
}
__device__ __forceinline__ void wait_async16() {  // current chunk done, next in flight
#if __has_builtin(__builtin_amdgcn_s_wait_asynccnt)
  __builtin_amdgcn_s_wait_asynccnt(16);
#else
  asm volatile("s_wait_asynccnt 0x10" ::: "memory");
#endif
  asm volatile("" ::: "memory");
}
__device__ __forceinline__ void wait_async0() {
#if __has_builtin(__builtin_amdgcn_s_wait_asynccnt)
  __builtin_amdgcn_s_wait_asynccnt(0);
#else
  asm volatile("s_wait_asynccnt 0x0" ::: "memory");
#endif
  asm volatile("" ::: "memory");
}

// ---------------------------------------------------------------------------
// Weight repack: Wt[col][k] = (bf16) W[k][col]   (K fixed at 512 here)
// Makes GEMM B-fragments two contiguous 16B loads instead of 16 strided b32.
// ---------------------------------------------------------------------------
__global__ void pack_w_kernel(const float* __restrict__ W, __bf16* __restrict__ Wt,
                              int Ncol) {
  const int t = blockIdx.x * blockDim.x + threadIdx.x;  // Ncol*512 threads
  const int col = t >> 9;
  const int k = t & 511;
  Wt[(size_t)col * 512 + k] = (__bf16)W[(size_t)k * Ncol + col];
}

// ---------------------------------------------------------------------------
// GEMM: C[M,N] = A[M,K](f32) * Wt[N,K](bf16, pre-transposed) + bias[N]
// One wave computes a 16x64 strip (4 C tiles), reusing each A fragment 4x.
// ---------------------------------------------------------------------------
__global__ void gemm16x64_bias_kernel(const float* __restrict__ A,
                                      const __bf16* __restrict__ Wt,
                                      const float* __restrict__ bias,
                                      float* __restrict__ C,
                                      int M, int Ncol, int K) {
  const int wave = threadIdx.x >> 5;
  const int lane = threadIdx.x & 31;
  const int m = lane & 15;
  const int g = lane >> 4;
  const int tilesN = Ncol >> 6;  // 64-wide strips
  const int tile = blockIdx.x * (blockDim.x >> 5) + wave;
  const int tm = tile / tilesN;
  const int tn = tile - tm * tilesN;
  if (tm >= (M >> 4)) return;  // uniform per wave
  const int cb = tn * 64;

  const float* arow = A + (size_t)(tm * 16 + m) * K;
  f32x8 acc[4];
#pragma unroll
  for (int j = 0; j < 4; ++j) acc[j] = zero8();

  for (int k0 = 0; k0 < K; k0 += 32) {
    f32x8 alo = *reinterpret_cast<const f32x8*>(arow + k0 + 8 * g);
    f32x8 ahi = *reinterpret_cast<const f32x8*>(arow + k0 + 16 + 8 * g);
    bf16x16 af;
#pragma unroll
    for (int e = 0; e < 8; ++e) { af[e] = (__bf16)alo[e]; af[8 + e] = (__bf16)ahi[e]; }
#pragma unroll
    for (int j = 0; j < 4; ++j) {
      bf16x16 bfr =
          frag_from_bf16_row(Wt + (size_t)(cb + j * 16 + m) * K, k0, g);
      acc[j] = wmma_bf16(af, bfr, acc[j]);
    }
  }
#pragma unroll
  for (int j = 0; j < 4; ++j) {
    const int col = cb + j * 16 + m;
    const float bv = bias[col];
#pragma unroll
    for (int r = 0; r < 8; ++r)
      C[(size_t)(tm * 16 + r + 8 * g) * Ncol + col] = acc[j][r] + bv;
  }
}

// ---------------------------------------------------------------------------
// RoPE (fp32 math before bf16 quantization) + pack.
//   blockIdx.y: 0 = Q (x 1/sqrt(HDIM)), 1 = K, 2 = V (transposed).
// Q,K: [n*NHD+h][l][d]   V: [n*NHD+h][d][l]
// ---------------------------------------------------------------------------
__global__ void rope_pack_kernel(const float* __restrict__ qkv,
                                 const int* __restrict__ pos_ids,
                                 __bf16* __restrict__ Qb,
                                 __bf16* __restrict__ Kb,
                                 __bf16* __restrict__ Vt) {
  const int t = blockIdx.x * blockDim.x + threadIdx.x;  // NB*NHD*LSEQ*HDIM
  const int d = t & (HDIM - 1);
  const int row = t >> 6;
  const int l = row & (LSEQ - 1);
  const int bh = row >> 11;
  const int h = bh & (NHD - 1);
  const int n = bh >> 3;
  const int which = blockIdx.y;
  const size_t src = ((size_t)n * LSEQ + l) * (3 * DMODEL) +
                     (size_t)which * DMODEL + (size_t)h * HDIM;
  if (which == 2) {
    Vt[((size_t)bh * HDIM + d) * LSEQ + l] = (__bf16)qkv[src + d];
    return;
  }
  const int i = d >> 1;
  const float inv_freq = __expf(-(float)i * (9.210340371976184f / 32.0f));
  const float ang = (float)pos_ids[(size_t)n * LSEQ + l] * inv_freq;
  float sv, cv;
  __sincosf(ang, &sv, &cv);
  const float e0 = qkv[src + (d & ~1)];
  const float e1 = qkv[src + (d | 1)];
  const float r = (d & 1) ? (e0 * sv + e1 * cv) : (e0 * cv - e1 * sv);
  if (which == 0)
    Qb[((size_t)bh * LSEQ + l) * HDIM + d] = (__bf16)(r * 0.125f);
  else
    Kb[((size_t)bh * LSEQ + l) * HDIM + d] = (__bf16)r;
}

// ---------------------------------------------------------------------------
// Causal flash attention. grid (NB*NHD, LSEQ/32), block 64 (2 waves).
// Each wave owns 16 Q rows and double-buffers 32-key K/V chunks into LDS with
// GLOBAL_LOAD_ASYNC_TO_LDS_B128, overlapping HBM latency with WMMA+softmax.
// ---------------------------------------------------------------------------
__global__ void flash_attn_kernel(const __bf16* __restrict__ Qb,
                                  const __bf16* __restrict__ Kb,
                                  const __bf16* __restrict__ Vt,
                                  const unsigned char* __restrict__ pad,
                                  float* __restrict__ attn) {
  __shared__ __align__(16) __bf16 Kst[2][2][32][64];  // [wave][buf][key][dim] 16KB
  __shared__ __align__(16) __bf16 Vst[2][2][64][32];  // [wave][buf][dim][key] 16KB
  __shared__ __align__(16) __bf16 Pst[2][16][40];     // P relayout slab 2.5KB

  const int wave = threadIdx.x >> 5;
  const int lane = threadIdx.x & 31;
  const int m = lane & 15;
  const int g = lane >> 4;
  const int bh = blockIdx.x;
  const int n = bh >> 3;
  const int h = bh & 7;
  const int q0 = blockIdx.y * 32 + wave * 16;

  const __bf16* Qh = Qb + (size_t)bh * LSEQ * HDIM;
  const __bf16* Kh = Kb + (size_t)bh * LSEQ * HDIM;
  const __bf16* Vh = Vt + (size_t)bh * HDIM * LSEQ;
  const unsigned char* padr = pad + (size_t)n * LSEQ;

  // Issue async prefetch of a 32-key chunk (16 x b128 per wave: 4KB K + 4KB V).
  auto prefetch = [&](int buf, int kbase) {
    __bf16* kb = &Kst[wave][buf][0][0];
    __bf16* vb = &Vst[wave][buf][0][0];
#pragma unroll
    for (int i = 0; i < 8; ++i) {
      const int idx = i * 32 + lane;  // 0..255 16B segments
      async_cp_b128(Kh + (size_t)(kbase + (idx >> 3)) * HDIM + (idx & 7) * 8,
                    kb + idx * 8);
      async_cp_b128(Vh + (size_t)(idx >> 2) * LSEQ + kbase + (idx & 3) * 8,
                    vb + idx * 8);
    }
  };

  const int nch = (q0 + 16 + 31) >> 5;  // chunks covering keys 0..q0+15
  prefetch(0, 0);

  // Q fragments (scale pre-folded into Q at pack time)
  bf16x16 qa[2];
#pragma unroll
  for (int t = 0; t < 2; ++t)
    qa[t] = frag_from_bf16_row(Qh + (size_t)(q0 + m) * HDIM, t * 32, g);

  f32x8 o[4];
#pragma unroll
  for (int t2 = 0; t2 < 4; ++t2) o[t2] = zero8();
  float mi[8], li[8];
#pragma unroll
  for (int r = 0; r < 8; ++r) { mi[r] = -INFINITY; li[r] = 0.f; }

  for (int kc = 0; kc < nch; ++kc) {
    const int kbase = kc << 5;
    const int cur = kc & 1;
    const bool more = (kc + 1) < nch;  // wave-uniform
    if (more) prefetch(1 - cur, (kc + 1) << 5);
    if (more) wait_async16(); else wait_async0();  // current chunk resident

    const __bf16* kbuf = &Kst[wave][cur][0][0];
    const __bf16* vbuf = &Vst[wave][cur][0][0];

    // ---- S = Q K^T for 2 x 16 keys (C layout: lane = col m, rows r+8g)
    f32x8 s[2];
#pragma unroll
    for (int t = 0; t < 2; ++t) {
      const int key = kbase + t * 16 + m;
      f32x8 accs = zero8();
#pragma unroll
      for (int kk = 0; kk < 2; ++kk) {
        bf16x16 bfr = frag_from_bf16_row(kbuf + (t * 16 + m) * HDIM, kk * 32, g);
        accs = wmma_bf16(qa[kk], bfr, accs);
      }
      const bool kill = (padr[key] != 0);
#pragma unroll
      for (int r = 0; r < 8; ++r) {
        const int qrow = q0 + r + 8 * g;
        s[t][r] = (kill || key > qrow) ? -INFINITY : accs[r];
      }
    }

    // ---- online softmax (xor masks 1,2,4,8 reduce within the 16-lane half)
    float alpha[8];
#pragma unroll
    for (int r = 0; r < 8; ++r) {
      float cm = fmaxf(s[0][r], s[1][r]);
#pragma unroll
      for (int off = 1; off < 16; off <<= 1)
        cm = fmaxf(cm, __shfl_xor(cm, off, 32));
      const float mnew = fmaxf(mi[r], cm);
      const float mg = (mnew == -INFINITY) ? 0.f : mnew;  // fully-masked guard
      alpha[r] = __expf(mi[r] - mg);
      mi[r] = mnew;
      const float p0 = __expf(s[0][r] - mg);
      const float p1 = __expf(s[1][r] - mg);
      s[0][r] = p0;
      s[1][r] = p1;
      float rs = p0 + p1;
#pragma unroll
      for (int off = 1; off < 16; off <<= 1)
        rs += __shfl_xor(rs, off, 32);
      li[r] = li[r] * alpha[r] + rs;
#pragma unroll
      for (int t2 = 0; t2 < 4; ++t2) o[t2][r] *= alpha[r];
    }

    // ---- C-frag -> A-frag relayout of P via per-wave LDS slab
#pragma unroll
    for (int t = 0; t < 2; ++t)
#pragma unroll
      for (int r = 0; r < 8; ++r)
        Pst[wave][r + 8 * g][t * 16 + m] = (__bf16)s[t][r];
    __asm__ volatile("s_wait_dscnt 0x0" ::: "memory");  // per-wave fence
    bf16x16 pa = frag_from_bf16_row(&Pst[wave][m][0], 0, g);

    // ---- O += P V (4 head-dim tiles from the staged V^T chunk)
#pragma unroll
    for (int t2 = 0; t2 < 4; ++t2) {
      bf16x16 vb = frag_from_bf16_row(vbuf + (t2 * 16 + m) * 32, 0, g);
      o[t2] = wmma_bf16(pa, vb, o[t2]);
    }
  }

  // ---- epilogue: normalize, scatter into [n][l][DMODEL]
#pragma unroll
  for (int r = 0; r < 8; ++r) {
    const float inv = (li[r] > 0.f) ? 1.f / li[r] : 0.f;  // nan_to_num semantics
    const size_t orow =
        ((size_t)n * LSEQ + q0 + r + 8 * g) * DMODEL + (size_t)h * HDIM;
#pragma unroll
    for (int t2 = 0; t2 < 4; ++t2) attn[orow + t2 * 16 + m] = o[t2][r] * inv;
  }
}

// ---------------------------------------------------------------------------
extern "C" void kernel_launch(void* const* d_in, const int* in_sizes, int n_in,
                              void* d_out, int out_size, void* d_ws, size_t ws_size,
                              hipStream_t stream) {
  (void)in_sizes; (void)n_in; (void)out_size; (void)ws_size;

  const float*         x    = (const float*)d_in[0];
  const int*           pos  = (const int*)d_in[1];
  const unsigned char* padm = (const unsigned char*)d_in[2];  // jnp bool -> u8
  const float*         Wqkv = (const float*)d_in[3];
  const float*         bqkv = (const float*)d_in[4];
  const float*         Wout = (const float*)d_in[5];
  const float*         bout = (const float*)d_in[6];
  float*               out  = (float*)d_out;

  char* ws = (char*)d_ws;
  float*  qkv   = (float*)ws;                                  // 48 MB
  __bf16* Qb    = (__bf16*)(ws + ((size_t)48 << 20));          // 8 MB
  __bf16* Kb    = (__bf16*)(ws + ((size_t)56 << 20));          // 8 MB
  __bf16* Vt    = (__bf16*)(ws + ((size_t)64 << 20));          // 8 MB
  __bf16* WtQKV = (__bf16*)(ws + ((size_t)72 << 20));          // 1.5 MB
  __bf16* WtOut = (__bf16*)(ws + ((size_t)74 << 20));          // 0.5 MB
  float*  attn  = (float*)ws;  // reuses qkv region (dead after rope_pack)

  const int Mrows = NB * LSEQ;  // 8192

  // 0) repack weights to bf16 [col][k]
  pack_w_kernel<<<(3 * DMODEL * DMODEL) / 256, 256, 0, stream>>>(Wqkv, WtQKV,
                                                                 3 * DMODEL);
  pack_w_kernel<<<(DMODEL * DMODEL) / 256, 256, 0, stream>>>(Wout, WtOut, DMODEL);

  // 1) qkv = x @ Wqkv + bqkv   (16x64 strips, 4 waves/block)
  {
    const int tiles = (Mrows / 16) * (3 * DMODEL / 64);  // 12288 wave-strips
    gemm16x64_bias_kernel<<<tiles / 4, 128, 0, stream>>>(x, WtQKV, bqkv, qkv,
                                                         Mrows, 3 * DMODEL, DMODEL);
  }
  // 2) RoPE + pack Q,K (bf16) and V (bf16 transposed)
  {
    const int total = NB * NHD * LSEQ * HDIM;
    rope_pack_kernel<<<dim3(total / 256, 3), 256, 0, stream>>>(qkv, pos, Qb, Kb, Vt);
  }
  // 3) causal flash attention with async double-buffered K/V staging
  flash_attn_kernel<<<dim3(NB * NHD, LSEQ / 32), 64, 0, stream>>>(Qb, Kb, Vt,
                                                                  padm, attn);
  // 4) out = attn @ Wout + bout
  {
    const int tiles = (Mrows / 16) * (DMODEL / 64);  // 4096 wave-strips
    gemm16x64_bias_kernel<<<tiles / 4, 128, 0, stream>>>(attn, WtOut, bout, out,
                                                         Mrows, DMODEL, DMODEL);
  }
}
